// MultilayerGRU_9801115369475
// MI455X (gfx1250) — matrix operand order repaired and checked by reading, not compiled, not run on hardware
//
#include <hip/hip_runtime.h>
#include <hip/hip_bf16.h>

#define DEV __device__ __forceinline__

typedef __attribute__((ext_vector_type(16))) __bf16 v16bf;
typedef __attribute__((ext_vector_type(8)))  __bf16 v8bf;
typedef __attribute__((ext_vector_type(8)))  float  v8f;

constexpr int B_ = 64, S_ = 512, I_ = 512, H_ = 512, L_ = 2, O_ = 512;

// ---------------- WMMA fragment helpers (CDNA5 bf16 16x16x32) ----------------
// A: row-major [M][K] bf16. Per ISA 7.12.2: lane<16 -> row M=lane, K = k0+0..7 and k0+16..23
//                                            lane>=16 -> row M=lane-16, K = k0+8..15 and k0+24..31
DEV v16bf load_a_frag(const __bf16* __restrict__ A, int lda, int m0, int k0, int lane) {
  int r = lane & 15, hi = lane >> 4;
  const __bf16* p = A + (size_t)(m0 + r) * lda + k0 + hi * 8;
  v8bf lo  = *(const v8bf*)p;         // K = k0 + hi*8 + 0..7
  v8bf hi8 = *(const v8bf*)(p + 16);  // K = k0 + 16 + hi*8 + 0..7
  return __builtin_shufflevector(lo, hi8, 0,1,2,3,4,5,6,7,8,9,10,11,12,13,14,15);
}

// B: stored pre-transposed as [K][N] bf16 (so B[k][n] = W[n][k]).
// Per ISA: lane L holds column N = n0 + L%16, K = k0 + (L/16)*8 + 0..7 and k0 + 16 + (L/16)*8 + 0..7.
DEV v16bf load_b_frag(const __bf16* __restrict__ Bt, int ldb, int k0, int n0, int lane) {
  const __bf16* p = Bt + (size_t)(k0 + (lane >> 4) * 8) * ldb + n0 + (lane & 15);
  v16bf b;
#pragma unroll
  for (int s = 0; s < 8; ++s) { b[s] = p[(size_t)s * ldb]; b[s + 8] = p[(size_t)(s + 16) * ldb]; }
  return b;
}

DEV v8f wmma_bf16(v16bf a, v16bf b, v8f c) {
  return __builtin_amdgcn_wmma_f32_16x16x32_bf16(false, a, false, b, (short)0, c, false, false);
}

DEV float sigmoidf_(float x) { return 1.0f / (1.0f + __expf(-x)); }
DEV float tanhf_(float x) {
  x = fminf(fmaxf(x, -15.0f), 15.0f);
  float e = __expf(2.0f * x);
  return (e - 1.0f) / (e + 1.0f);
}

// ---------------- setup kernels ----------------
__global__ void k_cvt_bf16(const float* __restrict__ src, __bf16* __restrict__ dst, int n) {
  int i = blockIdx.x * blockDim.x + threadIdx.x;
  if (i < n) dst[i] = (__bf16)src[i];
}

// W: [rows][cols] fp32 -> Wt: [cols][rows] bf16
__global__ void k_transpose_cvt(const float* __restrict__ W, __bf16* __restrict__ Wt,
                                int rows, int cols) {
  int i = blockIdx.x * blockDim.x + threadIdx.x;
  if (i < rows * cols) {
    int r = i / cols, c = i % cols;
    Wt[(size_t)c * rows + r] = (__bf16)W[i];
  }
}

__global__ void k_init_h(const float* __restrict__ hs, float* __restrict__ h0f,
                         float* __restrict__ h1f, __bf16* __restrict__ h0b,
                         __bf16* __restrict__ h1b) {
  int i = blockIdx.x * blockDim.x + threadIdx.x;  // over B*H
  if (i < B_ * H_) {
    int b = i / H_, h = i % H_;
    float v0 = hs[(size_t)b * L_ * H_ + h];
    float v1 = hs[(size_t)b * L_ * H_ + H_ + h];
    h0f[i] = v0; h1f[i] = v1;
    h0b[i] = (__bf16)v0; h1b[i] = (__bf16)v1;
  }
}

// ---------------- phase 1: bulk layer-0 input projections ----------------
// X: [B*S, I] bf16 (row m = (b, t) with m = b*S + t); W*T: [I, H] bf16.
// Outputs Xz/Xr/Xg: [B*S, H] fp32 preactivations (no bias).
// Register blocking: each wave computes TWO M-tiles (32 rows) x 16 cols for all
// 3 gates -> every B fragment feeds 2 WMMAs, every A fragment feeds 3 WMMAs.
__global__ void k_xproj(const __bf16* __restrict__ X,
                        const __bf16* __restrict__ WzT, const __bf16* __restrict__ WrT,
                        const __bf16* __restrict__ WgT,
                        float* __restrict__ Xz, float* __restrict__ Xr, float* __restrict__ Xg) {
  int lane = threadIdx.x & 31;
  int wave = threadIdx.x >> 5;
  int m0 = blockIdx.x * 32;
  int n0 = (blockIdx.y * 4 + wave) * 16;
  v8f az0 = {}, ar0 = {}, ag0 = {};
  v8f az1 = {}, ar1 = {}, ag1 = {};
  for (int k0 = 0; k0 < I_; k0 += 32) {
    v16bf a0 = load_a_frag(X, I_, m0,      k0, lane);
    v16bf a1 = load_a_frag(X, I_, m0 + 16, k0, lane);
    v16bf fz = load_b_frag(WzT, H_, k0, n0, lane);
    v16bf fr = load_b_frag(WrT, H_, k0, n0, lane);
    v16bf fg = load_b_frag(WgT, H_, k0, n0, lane);
    az0 = wmma_bf16(a0, fz, az0);  az1 = wmma_bf16(a1, fz, az1);
    ar0 = wmma_bf16(a0, fr, ar0);  ar1 = wmma_bf16(a1, fr, ar1);
    ag0 = wmma_bf16(a0, fg, ag0);  ag1 = wmma_bf16(a1, fg, ag1);
  }
  int r = lane & 15, hi = lane >> 4;
#pragma unroll
  for (int v = 0; v < 8; ++v) {
    size_t i0 = (size_t)(m0 + v + 8 * hi) * H_ + n0 + r;
    size_t i1 = (size_t)(m0 + 16 + v + 8 * hi) * H_ + n0 + r;
    Xz[i0] = az0[v]; Xr[i0] = ar0[v]; Xg[i0] = ag0[v];
    Xz[i1] = az1[v]; Xr[i1] = ar1[v]; Xg[i1] = ag1[v];
  }
}

// ---------------- per-timestep kernels (grid (B/16, H/16), block 32) ----------------
// Layer-0 stage A: z,r gates + r*h (x-parts come precomputed).
__global__ void k_gru_a_l0(const __bf16* __restrict__ hb, const float* __restrict__ hf,
                           const __bf16* __restrict__ WzhT, const __bf16* __restrict__ WrhT,
                           const float* __restrict__ Xz, const float* __restrict__ Xr,
                           const float* __restrict__ bz, const float* __restrict__ br,
                           int t, float* __restrict__ zbuf, __bf16* __restrict__ rhb) {
  int lane = threadIdx.x & 31;
  int m0 = blockIdx.x * 16, n0 = blockIdx.y * 16;
  v8f az = {}, ar = {};
  for (int k0 = 0; k0 < H_; k0 += 32) {
    v16bf a = load_a_frag(hb, H_, m0, k0, lane);
    az = wmma_bf16(a, load_b_frag(WzhT, H_, k0, n0, lane), az);
    ar = wmma_bf16(a, load_b_frag(WrhT, H_, k0, n0, lane), ar);
  }
  int r = lane & 15, hi = lane >> 4;
#pragma unroll
  for (int v = 0; v < 8; ++v) {
    int m = m0 + v + 8 * hi, n = n0 + r;
    size_t xi = ((size_t)m * S_ + t) * H_ + n;
    size_t i  = (size_t)m * H_ + n;
    float z  = sigmoidf_(Xz[xi] + az[v] + bz[n]);
    float rr = sigmoidf_(Xr[xi] + ar[v] + br[n]);
    zbuf[i] = z;
    rhb[i]  = (__bf16)(rr * hf[i]);
  }
}

// Layer-1 stage A: x-projections from h0 (5 GEMM chains) + z,r gates + r*h; stash g's x-part.
__global__ void k_gru_a_l1(const __bf16* __restrict__ xb, const __bf16* __restrict__ hb,
                           const float* __restrict__ hf,
                           const __bf16* __restrict__ WzxT, const __bf16* __restrict__ WrxT,
                           const __bf16* __restrict__ WgxT,
                           const __bf16* __restrict__ WzhT, const __bf16* __restrict__ WrhT,
                           const float* __restrict__ bz, const float* __restrict__ br,
                           float* __restrict__ zbuf, __bf16* __restrict__ rhb,
                           float* __restrict__ xg) {
  int lane = threadIdx.x & 31;
  int m0 = blockIdx.x * 16, n0 = blockIdx.y * 16;
  v8f axz = {}, axr = {}, axg = {}, ahz = {}, ahr = {};
  for (int k0 = 0; k0 < H_; k0 += 32) {
    v16bf ax = load_a_frag(xb, H_, m0, k0, lane);
    v16bf ah = load_a_frag(hb, H_, m0, k0, lane);
    axz = wmma_bf16(ax, load_b_frag(WzxT, H_, k0, n0, lane), axz);
    axr = wmma_bf16(ax, load_b_frag(WrxT, H_, k0, n0, lane), axr);
    axg = wmma_bf16(ax, load_b_frag(WgxT, H_, k0, n0, lane), axg);
    ahz = wmma_bf16(ah, load_b_frag(WzhT, H_, k0, n0, lane), ahz);
    ahr = wmma_bf16(ah, load_b_frag(WrhT, H_, k0, n0, lane), ahr);
  }
  int r = lane & 15, hi = lane >> 4;
#pragma unroll
  for (int v = 0; v < 8; ++v) {
    int m = m0 + v + 8 * hi, n = n0 + r;
    size_t i = (size_t)m * H_ + n;
    float z  = sigmoidf_(axz[v] + ahz[v] + bz[n]);
    float rr = sigmoidf_(axr[v] + ahr[v] + br[n]);
    zbuf[i] = z;
    rhb[i]  = (__bf16)(rr * hf[i]);
    xg[i]   = axg[v];
  }
}

// Stage B (both layers): g = tanh(Xg + (r*h) @ Wgh^T + bg); h = z*h + (1-z)*g.
__global__ void k_gru_b(const __bf16* __restrict__ rhb, const __bf16* __restrict__ WghT,
                        const float* __restrict__ Xg, size_t xgRowStride, size_t xgOff,
                        const float* __restrict__ bg, const float* __restrict__ zbuf,
                        float* __restrict__ hf, __bf16* __restrict__ hb) {
  int lane = threadIdx.x & 31;
  int m0 = blockIdx.x * 16, n0 = blockIdx.y * 16;
  v8f ag = {};
  for (int k0 = 0; k0 < H_; k0 += 32) {
    v16bf a = load_a_frag(rhb, H_, m0, k0, lane);
    ag = wmma_bf16(a, load_b_frag(WghT, H_, k0, n0, lane), ag);
  }
  int r = lane & 15, hi = lane >> 4;
#pragma unroll
  for (int v = 0; v < 8; ++v) {
    int m = m0 + v + 8 * hi, n = n0 + r;
    size_t i = (size_t)m * H_ + n;
    float g = tanhf_(Xg[(size_t)m * xgRowStride + xgOff + n] + ag[v] + bg[n]);
    float z = zbuf[i];
    float hnew = z * hf[i] + (1.0f - z) * g;
    hf[i] = hnew;
    hb[i] = (__bf16)hnew;
  }
}

// Output projection: y_t = h1 @ Wout^T + bout -> layer_output[b][t][o]
__global__ void k_outproj(const __bf16* __restrict__ hb, const __bf16* __restrict__ WoT,
                          const float* __restrict__ bo, int t, float* __restrict__ out) {
  int lane = threadIdx.x & 31;
  int m0 = blockIdx.x * 16, n0 = blockIdx.y * 16;
  v8f acc = {};
  for (int k0 = 0; k0 < H_; k0 += 32) {
    v16bf a = load_a_frag(hb, H_, m0, k0, lane);
    acc = wmma_bf16(a, load_b_frag(WoT, O_, k0, n0, lane), acc);
  }
  int r = lane & 15, hi = lane >> 4;
#pragma unroll
  for (int v = 0; v < 8; ++v) {
    int m = m0 + v + 8 * hi, n = n0 + r;
    out[((size_t)m * S_ + t) * O_ + n] = acc[v] + bo[n];
  }
}

__global__ void k_final(const float* __restrict__ h0f, const float* __restrict__ h1f,
                        float* __restrict__ out) {
  int i = blockIdx.x * blockDim.x + threadIdx.x;  // over B*H
  if (i < B_ * H_) {
    int b = i / H_, h = i % H_;
    size_t base = (size_t)B_ * S_ * O_ + (size_t)b * L_ * H_;
    out[base + h]      = h0f[i];
    out[base + H_ + h] = h1f[i];
  }
}

// ---------------- host side ----------------
extern "C" void kernel_launch(void* const* d_in, const int* in_sizes, int n_in,
                              void* d_out, int out_size, void* d_ws, size_t ws_size,
                              hipStream_t stream) {
  (void)in_sizes; (void)n_in; (void)out_size; (void)ws_size;
  const float* input  = (const float*)d_in[0];   // [B,S,I]
  const float* hidden = (const float*)d_in[1];   // [B,L,H]
  const float* Wzx = (const float*)d_in[2];      // [L,H,I]
  const float* Wzh = (const float*)d_in[3];      // [L,H,H]
  const float* bz  = (const float*)d_in[4];      // [L,H]
  const float* Wrx = (const float*)d_in[5];
  const float* Wrh = (const float*)d_in[6];
  const float* br  = (const float*)d_in[7];
  const float* Wgx = (const float*)d_in[8];
  const float* Wgh = (const float*)d_in[9];
  const float* bg  = (const float*)d_in[10];
  const float* Wout= (const float*)d_in[11];     // [O,H]
  const float* bout= (const float*)d_in[12];     // [O]
  float* out = (float*)d_out;

  // workspace carve (256B aligned chunks)
  char* ws = (char*)d_ws;
  size_t off = 0;
  auto carve = [&](size_t bytes) -> void* {
    void* p = ws + off;
    off += (bytes + 255) & ~(size_t)255;
    return p;
  };
  __bf16* inb  = (__bf16*)carve((size_t)B_ * S_ * I_ * 2);
  __bf16* WzxT = (__bf16*)carve((size_t)L_ * I_ * H_ * 2);
  __bf16* WzhT = (__bf16*)carve((size_t)L_ * H_ * H_ * 2);
  __bf16* WrxT = (__bf16*)carve((size_t)L_ * I_ * H_ * 2);
  __bf16* WrhT = (__bf16*)carve((size_t)L_ * H_ * H_ * 2);
  __bf16* WgxT = (__bf16*)carve((size_t)L_ * I_ * H_ * 2);
  __bf16* WghT = (__bf16*)carve((size_t)L_ * H_ * H_ * 2);
  __bf16* WoT  = (__bf16*)carve((size_t)H_ * O_ * 2);
  float*  Xz   = (float*)carve((size_t)B_ * S_ * H_ * 4);
  float*  Xr   = (float*)carve((size_t)B_ * S_ * H_ * 4);
  float*  Xg   = (float*)carve((size_t)B_ * S_ * H_ * 4);
  float*  h0f  = (float*)carve((size_t)B_ * H_ * 4);
  float*  h1f  = (float*)carve((size_t)B_ * H_ * 4);
  __bf16* h0b  = (__bf16*)carve((size_t)B_ * H_ * 2);
  __bf16* h1b  = (__bf16*)carve((size_t)B_ * H_ * 2);
  __bf16* rhb  = (__bf16*)carve((size_t)B_ * H_ * 2);
  float*  zbuf = (float*)carve((size_t)B_ * H_ * 4);
  float*  xg1  = (float*)carve((size_t)B_ * H_ * 4);

  // --- setup: convert input + transpose-convert weights to bf16 ---
  {
    int n = B_ * S_ * I_;
    k_cvt_bf16<<<(n + 255) / 256, 256, 0, stream>>>(input, inb, n);
  }
  const int wBlocks = (H_ * I_ + 255) / 256;
  for (int l = 0; l < L_; ++l) {
    k_transpose_cvt<<<wBlocks, 256, 0, stream>>>(Wzx + (size_t)l * H_ * I_, WzxT + (size_t)l * I_ * H_, H_, I_);
    k_transpose_cvt<<<wBlocks, 256, 0, stream>>>(Wzh + (size_t)l * H_ * H_, WzhT + (size_t)l * H_ * H_, H_, H_);
    k_transpose_cvt<<<wBlocks, 256, 0, stream>>>(Wrx + (size_t)l * H_ * I_, WrxT + (size_t)l * I_ * H_, H_, I_);
    k_transpose_cvt<<<wBlocks, 256, 0, stream>>>(Wrh + (size_t)l * H_ * H_, WrhT + (size_t)l * H_ * H_, H_, H_);
    k_transpose_cvt<<<wBlocks, 256, 0, stream>>>(Wgx + (size_t)l * H_ * I_, WgxT + (size_t)l * I_ * H_, H_, I_);
    k_transpose_cvt<<<wBlocks, 256, 0, stream>>>(Wgh + (size_t)l * H_ * H_, WghT + (size_t)l * H_ * H_, H_, H_);
  }
  k_transpose_cvt<<<(O_ * H_ + 255) / 256, 256, 0, stream>>>(Wout, WoT, O_, H_);
  k_init_h<<<(B_ * H_ + 255) / 256, 256, 0, stream>>>(hidden, h0f, h1f, h0b, h1b);

  // --- phase 1: bulk layer-0 input projections (one big WMMA GEMM) ---
  {
    dim3 grid((B_ * S_) / 32, H_ / 64);  // 1024 x 8; 4 waves/block, 32x16 per gate per wave
    k_xproj<<<grid, 128, 0, stream>>>(inb, WzxT, WrxT, WgxT, Xz, Xr, Xg);
  }

  // --- phase 2: sequential scan (5 small WMMA kernels per timestep) ---
  dim3 gG(B_ / 16, H_ / 16);  // (4, 32), one wave per 16x16 tile
  for (int t = 0; t < S_; ++t) {
    k_gru_a_l0<<<gG, 32, 0, stream>>>(h0b, h0f, WzhT, WrhT, Xz, Xr, bz, br, t, zbuf, rhb);
    k_gru_b  <<<gG, 32, 0, stream>>>(rhb, WghT, Xg, (size_t)S_ * H_, (size_t)t * H_, bg, zbuf, h0f, h0b);
    k_gru_a_l1<<<gG, 32, 0, stream>>>(h0b, h1b, h1f,
                                      WzxT + (size_t)I_ * H_, WrxT + (size_t)I_ * H_, WgxT + (size_t)I_ * H_,
                                      WzhT + (size_t)H_ * H_, WrhT + (size_t)H_ * H_,
                                      bz + H_, br + H_, zbuf, rhb, xg1);
    k_gru_b  <<<gG, 32, 0, stream>>>(rhb, WghT + (size_t)H_ * H_, xg1, (size_t)H_, 0, bg + H_, zbuf, h1f, h1b);
    k_outproj<<<gG, 32, 0, stream>>>(h1b, WoT, bout, t, out);
  }

  // --- final hidden state ---
  k_final<<<(B_ * H_ + 255) / 256, 256, 0, stream>>>(h0f, h1f, out);
}
